// RSSM_82154134438727
// MI455X (gfx1250) — matrix operand order repaired and checked by reading, not compiled, not run on hardware
//
#include <hip/hip_runtime.h>
#include <hip/hip_bf16.h>
#include <math.h>

// MI455X / gfx1250, wave32. Core GEMMs use v_wmma_f32_16x16x32_bf16.
typedef __bf16 bf16_t;
typedef bf16_t v16bf __attribute__((ext_vector_type(16)));
typedef bf16_t v8bf  __attribute__((ext_vector_type(8)));
typedef float  v8f   __attribute__((ext_vector_type(8)));

#define B_   256
#define T_   64
#define RECS 2048

// ---------------- tiled bf16 WMMA GEMM ----------------
// C[M,N] = act( A[M,K](f32) * Bt[N,K](bf16)^T + bias ),  act: 0=none, 1=ELU
// Optional fused normalization of A (encoder path): a = (a - nmean[g]) / nstd[g], g = k>>9.
#define BM 64
#define BN 128
#define BK 32
#define SA 56   // LDS row stride in bf16 (112B: 16B-aligned, conflict-free 16-row access)
#define SB 56

__global__ __launch_bounds__(256) void gemm_bf16_kernel(
    const float* __restrict__ A, int lda,
    const bf16_t* __restrict__ Bt,
    const float* __restrict__ bias,
    float* __restrict__ C,
    int M, int N, int K, int act,
    const float* __restrict__ nmean, const float* __restrict__ nstd)
{
  __shared__ bf16_t sA[BM * SA];
  __shared__ bf16_t sB[BN * SB];

  const int tid  = threadIdx.x;
  const int wave = tid >> 5;
  const int lane = tid & 31;
  const int wm = wave >> 2;           // 0..1 : 32-row slab
  const int wn = wave & 3;            // 0..3 : 32-col slab
  const int m0 = blockIdx.y * BM;
  const int n0 = blockIdx.x * BN;

  const int arow = tid >> 2, acg = tid & 3;   // A staging: 8 f32 / thread
  const int brow = tid >> 1, bh  = tid & 1;   // B staging: 16 bf16 / thread
  const int fm  = lane & 15;                  // fragment row (A: M, B: N)
  const int fks = lane >> 4;                  // K-half selector

  // Hoist normalization constants: only 4 groups; do the divisions ONCE,
  // select in-loop with a 2-level cndmask chain (no per-iteration v_div/s_load).
  const bool do_norm = (nmean != nullptr);
  float mu0 = 0.f, mu1 = 0.f, mu2 = 0.f, mu3 = 0.f;
  float iv0 = 1.f, iv1 = 1.f, iv2 = 1.f, iv3 = 1.f;
  if (do_norm) {
    mu0 = nmean[0]; mu1 = nmean[1]; mu2 = nmean[2]; mu3 = nmean[3];
    iv0 = 1.0f / nstd[0]; iv1 = 1.0f / nstd[1];
    iv2 = 1.0f / nstd[2]; iv3 = 1.0f / nstd[3];
  }

  v8f acc[2][2];
  #pragma unroll
  for (int i = 0; i < 2; ++i)
    #pragma unroll
    for (int j = 0; j < 2; ++j)
      #pragma unroll
      for (int r = 0; r < 8; ++r) acc[i][j][r] = 0.0f;

  for (int kk = 0; kk < K; kk += BK) {
    __syncthreads();
    // ---- stage A tile: f32 -> bf16, optional fused normalize ----
    {
      const float* src = A + (size_t)(m0 + arow) * lda + kk + acg * 8;
      float4 f0 = ((const float4*)src)[0];
      float4 f1 = ((const float4*)src)[1];
      if (do_norm) {
        int g = (kk + acg * 8) >> 9;                       // 4 groups of 512
        float mu = (g & 2) ? ((g & 1) ? mu3 : mu2) : ((g & 1) ? mu1 : mu0);
        float iv = (g & 2) ? ((g & 1) ? iv3 : iv2) : ((g & 1) ? iv1 : iv0);
        f0.x=(f0.x-mu)*iv; f0.y=(f0.y-mu)*iv; f0.z=(f0.z-mu)*iv; f0.w=(f0.w-mu)*iv;
        f1.x=(f1.x-mu)*iv; f1.y=(f1.y-mu)*iv; f1.z=(f1.z-mu)*iv; f1.w=(f1.w-mu)*iv;
      }
      v8bf v;
      v[0]=(bf16_t)f0.x; v[1]=(bf16_t)f0.y; v[2]=(bf16_t)f0.z; v[3]=(bf16_t)f0.w;
      v[4]=(bf16_t)f1.x; v[5]=(bf16_t)f1.y; v[6]=(bf16_t)f1.z; v[7]=(bf16_t)f1.w;
      *(v8bf*)&sA[arow * SA + acg * 8] = v;
    }
    // ---- stage B^T tile (bf16, contiguous along K) ----
    {
      const uint4* src = (const uint4*)(Bt + (size_t)(n0 + brow) * K + kk + bh * 16);
      uint4 u0 = src[0], u1 = src[1];
      uint4* d = (uint4*)&sB[brow * SB + bh * 16];
      d[0] = u0; d[1] = u1;
    }
    if (kk + BK < K) {   // global_prefetch_b8 of next K tile
      __builtin_prefetch(A  + (size_t)(m0 + arow) * lda + kk + BK + acg * 8, 0, 3);
      __builtin_prefetch(Bt + (size_t)(n0 + brow) * K   + kk + BK + bh * 16, 0, 3);
    }
    __syncthreads();

    // A fragment (16x32 bf16): lane holds row fm, K = fks*8..+7 and 16+fks*8..+7
    v16bf aF[2], bF[2];
    #pragma unroll
    for (int i = 0; i < 2; ++i) {
      int base = (wm * 32 + i * 16 + fm) * SA + fks * 8;
      v8bf lo = *(v8bf*)&sA[base];
      v8bf hi = *(v8bf*)&sA[base + 16];
      aF[i] = __builtin_shufflevector(lo, hi, 0,1,2,3,4,5,6,7,8,9,10,11,12,13,14,15);
    }
    // B fragment (32x16 bf16): lane holds col fm, K = fks*8..+7 and 16+fks*8..+7
    #pragma unroll
    for (int j = 0; j < 2; ++j) {
      int base = (wn * 32 + j * 16 + fm) * SB + fks * 8;
      v8bf lo = *(v8bf*)&sB[base];
      v8bf hi = *(v8bf*)&sB[base + 16];
      bF[j] = __builtin_shufflevector(lo, hi, 0,1,2,3,4,5,6,7,8,9,10,11,12,13,14,15);
    }
    #pragma unroll
    for (int i = 0; i < 2; ++i)
      #pragma unroll
      for (int j = 0; j < 2; ++j)
        acc[i][j] = __builtin_amdgcn_wmma_f32_16x16x32_bf16(
            false, aF[i], false, bF[j], (short)0, acc[i][j], false, false);
  }

  // ---- epilogue: D layout -> lane fm is N, rows fks*8 + r ----
  #pragma unroll
  for (int i = 0; i < 2; ++i)
    #pragma unroll
    for (int j = 0; j < 2; ++j) {
      int gn  = n0 + wn * 32 + j * 16 + fm;
      int gm0 = m0 + wm * 32 + i * 16 + fks * 8;
      float bv = bias ? bias[gn] : 0.0f;
      #pragma unroll
      for (int r = 0; r < 8; ++r) {
        float v = acc[i][j][r] + bv;
        if (act == 1) v = (v > 0.0f) ? v : expm1f(v);   // ELU
        C[(size_t)(gm0 + r) * N + gn] = v;
      }
    }
}

// ---------------- weight prep: f32 -> bf16, optional transpose, K-pad with zeros ----
__global__ void prep_kernel(const float* __restrict__ src, bf16_t* __restrict__ dst,
                            int N, int Ks, int Kp, int tr) {
  size_t i = (size_t)blockIdx.x * blockDim.x + threadIdx.x;
  size_t tot = (size_t)N * Kp;
  if (i >= tot) return;
  int n = (int)(i / Kp);
  int k = (int)(i % Kp);
  float v = 0.0f;
  if (k < Ks) v = tr ? src[(size_t)k * N + n] : src[(size_t)n * Ks + k];
  dst[i] = (bf16_t)v;
}

__global__ void zero_kernel(float* p, size_t n) {
  size_t i = (size_t)blockIdx.x * blockDim.x + threadIdx.x;
  if (i < n) p[i] = 0.0f;
}

// concat [s1 | s2 | zero-pad] per row
__global__ void concat2_kernel(float* __restrict__ dst, int wdst,
                               const float* __restrict__ s1, int w1, int ld1,
                               const float* __restrict__ s2, int w2, int ld2,
                               int rows) {
  int i = blockIdx.x * blockDim.x + threadIdx.x;
  if (i >= rows * wdst) return;
  int b = i / wdst, j = i - b * wdst;
  float v = 0.0f;
  if (j < w1)            v = s1[(size_t)b * ld1 + j];
  else if (j < w1 + w2)  v = s2[(size_t)b * ld2 + (j - w1)];
  dst[i] = v;
}

// GRU cell (PyTorch GRUCell math)
__global__ void gru_kernel(const float* __restrict__ gx, const float* __restrict__ gh,
                           const float* __restrict__ h, float* __restrict__ hn, int rows) {
  int i = blockIdx.x * blockDim.x + threadIdx.x;
  if (i >= rows * RECS) return;
  int b = i >> 11, j = i & 2047;
  const float* gxb = gx + (size_t)b * 3 * RECS;
  const float* ghb = gh + (size_t)b * 3 * RECS;
  float r = 1.0f / (1.0f + __expf(-(gxb[j]        + ghb[j])));
  float u = 1.0f / (1.0f + __expf(-(gxb[2048 + j] + ghb[2048 + j])));
  float n = tanhf(gxb[4096 + j] + r * ghb[4096 + j]);
  hn[i] = (1.0f - u) * n + u * h[i];
}

// straight-through sample == one_hot(argmax) in forward. One wave32 per (b,l); lane = class.
__global__ void latent_kernel(const float* __restrict__ post, float* __restrict__ lat, int B) {
  int lane = threadIdx.x & 31;
  int g = blockIdx.x * (blockDim.x >> 5) + (threadIdx.x >> 5);
  if (g >= B * 32) return;
  int b = g >> 5, l = g & 31;
  float v = post[(size_t)b * 1024 + l * 32 + lane];
  float bv = v; int bl = lane;
  #pragma unroll
  for (int m = 16; m; m >>= 1) {
    float ov = __shfl_xor(bv, m);
    int   ol = __shfl_xor(bl, m);
    if (ov > bv || (ov == bv && ol < bl)) { bv = ov; bl = ol; }
  }
  lat[(size_t)b * 1024 + l * 32 + lane] = (lane == bl) ? 1.0f : 0.0f;
}

// forward KL: accumulate max(KL(post||prior), FREE_NAT) per batch row. One wave per row.
__global__ void kl_kernel(const float* __restrict__ post, const float* __restrict__ prior,
                          float* __restrict__ acc, int B) {
  int lane = threadIdx.x & 31;
  int b = blockIdx.x * (blockDim.x >> 5) + (threadIdx.x >> 5);
  if (b >= B) return;
  float kl = 0.0f;
  for (int l = 0; l < 32; ++l) {
    float pl = post [(size_t)b * 1024 + l * 32 + lane];
    float ql = prior[(size_t)b * 1024 + l * 32 + lane];
    float mp = pl, mq = ql;
    #pragma unroll
    for (int m = 16; m; m >>= 1) { mp = fmaxf(mp, __shfl_xor(mp, m)); mq = fmaxf(mq, __shfl_xor(mq, m)); }
    float sp = __expf(pl - mp), sq = __expf(ql - mq);
    #pragma unroll
    for (int m = 16; m; m >>= 1) { sp += __shfl_xor(sp, m); sq += __shfl_xor(sq, m); }
    float logp = pl - mp - __logf(sp);
    float logq = ql - mq - __logf(sq);
    float term = __expf(logp) * (logp - logq);
    #pragma unroll
    for (int m = 16; m; m >>= 1) term += __shfl_xor(term, m);
    kl += term;
  }
  if (lane == 0) atomicAdd(acc, fmaxf(kl, 1.0f));   // 0.8*max + 0.2*max == max in fwd
}

// recon: sum (pred - norm(zs[:,t+1]))^2, one block-atomic per block
__global__ void recon_kernel(const float* __restrict__ pred, const float* __restrict__ zs,
                             const float* __restrict__ zm, const float* __restrict__ zsd,
                             float* __restrict__ acc, int t) {
  __shared__ float red[256];
  int tid = threadIdx.x;
  int i = blockIdx.x * 256 + tid;
  float v = 0.0f;
  if (i < B_ * 2048) {
    int b = i >> 11, j = i & 2047;
    int g = j >> 9;
    float zn = __fdividef(zs[((size_t)b * T_ + (t + 1)) * 2048 + j] - zm[g], zsd[g]);
    float d = pred[i] - zn;
    v = d * d;
  }
  red[tid] = v; __syncthreads();
  for (int s = 128; s > 0; s >>= 1) { if (tid < s) red[tid] += red[tid + s]; __syncthreads(); }
  if (tid == 0) atomicAdd(acc, red[0]);
}

__global__ void finalize_kernel(const float* racc, const float* kacc, float* out,
                                float rden, float kden) {
  if (threadIdx.x == 0 && blockIdx.x == 0) out[0] = racc[0] / rden + kacc[0] / kden;
}

// ---------------------------------------------------------------------------
extern "C" void kernel_launch(void* const* d_in, const int* in_sizes, int n_in,
                              void* d_out, int out_size, void* d_ws, size_t ws_size,
                              hipStream_t stream) {
  (void)in_sizes; (void)n_in; (void)out_size; (void)ws_size;
  const float* zs      = (const float*)d_in[0];
  const float* actions = (const float*)d_in[1];
  const float* z_mean  = (const float*)d_in[2];
  const float* z_std   = (const float*)d_in[3];
  const float* enc_W   = (const float*)d_in[4];
  const float* enc_b   = (const float*)d_in[5];
  const float* pre_W   = (const float*)d_in[6];
  const float* pre_b   = (const float*)d_in[7];
  const float* W_ih    = (const float*)d_in[8];
  const float* W_hh    = (const float*)d_in[9];
  const float* b_ih    = (const float*)d_in[10];
  const float* b_hh    = (const float*)d_in[11];
  const float* tW1     = (const float*)d_in[12];
  const float* tb1     = (const float*)d_in[13];
  const float* tW2     = (const float*)d_in[14];
  const float* tb2     = (const float*)d_in[15];
  const float* rW1     = (const float*)d_in[16];
  const float* rb1     = (const float*)d_in[17];
  const float* rW2     = (const float*)d_in[18];
  const float* rb2     = (const float*)d_in[19];
  const float* dW1     = (const float*)d_in[20];
  const float* db1     = (const float*)d_in[21];
  const float* dW2     = (const float*)d_in[22];
  const float* db2     = (const float*)d_in[23];

  char* ws = (char*)d_ws;
  size_t off = 0;
  auto alloc = [&](size_t bytes) -> void* {
    off = (off + 255) & ~(size_t)255;
    void* p = ws + off; off += bytes; return p;
  };

  // bf16 weights (B^T layout, N x Kp row-major; Kp multiple of 32)
  bf16_t* enc_Wt = (bf16_t*)alloc((size_t)1024 * 2048 * 2);
  bf16_t* pre_Wt = (bf16_t*)alloc((size_t)1024 * 1056 * 2);
  bf16_t* W_ihb  = (bf16_t*)alloc((size_t)6144 * 1024 * 2);
  bf16_t* W_hhb  = (bf16_t*)alloc((size_t)6144 * 2048 * 2);
  bf16_t* tW1t   = (bf16_t*)alloc((size_t)1024 * 2048 * 2);
  bf16_t* tW2t   = (bf16_t*)alloc((size_t)1024 * 1024 * 2);
  bf16_t* rW1t   = (bf16_t*)alloc((size_t)1024 * 3072 * 2);
  bf16_t* rW2t   = (bf16_t*)alloc((size_t)1024 * 1024 * 2);
  bf16_t* dW1t   = (bf16_t*)alloc((size_t)1024 * 3072 * 2);
  bf16_t* dW2t   = (bf16_t*)alloc((size_t)2048 * 1024 * 2);
  // f32 activations
  float* encoded = (float*)alloc((size_t)B_ * T_ * 1024 * 4);
  float* pre_in  = (float*)alloc((size_t)B_ * 1056 * 4);
  float* xbuf    = (float*)alloc((size_t)B_ * 1024 * 4);
  float* gxbuf   = (float*)alloc((size_t)B_ * 6144 * 4);
  float* ghbuf   = (float*)alloc((size_t)B_ * 6144 * 4);
  float* hping   = (float*)alloc((size_t)B_ * RECS * 4);
  float* hpong   = (float*)alloc((size_t)B_ * RECS * 4);
  float* lping   = (float*)alloc((size_t)B_ * 1024 * 4);
  float* lpong   = (float*)alloc((size_t)B_ * 1024 * 4);
  float* t1buf   = (float*)alloc((size_t)B_ * 1024 * 4);
  float* priorb  = (float*)alloc((size_t)B_ * 1024 * 4);
  float* post_in = (float*)alloc((size_t)B_ * 3072 * 4);
  float* r1buf   = (float*)alloc((size_t)B_ * 1024 * 4);
  float* postb   = (float*)alloc((size_t)B_ * 1024 * 4);
  float* dec_in  = (float*)alloc((size_t)B_ * 3072 * 4);
  float* d1buf   = (float*)alloc((size_t)B_ * 1024 * 4);
  float* predb   = (float*)alloc((size_t)B_ * 2048 * 4);
  float* accs    = (float*)alloc(256);   // [0]=recon sum, [1]=kl sum

  auto prep = [&](const float* s, bf16_t* d, int N, int Ks, int Kp, int tr) {
    size_t tot = (size_t)N * Kp;
    prep_kernel<<<dim3((unsigned)((tot + 255) / 256)), dim3(256), 0, stream>>>(s, d, N, Ks, Kp, tr);
  };
  auto gemm = [&](const float* A, int lda, const bf16_t* Bt, const float* bias, float* C,
                  int M, int N, int K, int act, const float* nm, const float* ns) {
    gemm_bf16_kernel<<<dim3(N / BN, M / BM), dim3(256), 0, stream>>>(
        A, lda, Bt, bias, C, M, N, K, act, nm, ns);
  };
  auto concat2 = [&](float* dst, int wd, const float* s1, int w1, int ld1,
                     const float* s2, int w2, int ld2) {
    int tot = B_ * wd;
    concat2_kernel<<<dim3((tot + 255) / 256), dim3(256), 0, stream>>>(dst, wd, s1, w1, ld1, s2, w2, ld2, B_);
  };

  // ---- weight conversion (every call; deterministic) ----
  prep(enc_W, enc_Wt, 1024, 2048, 2048, 1);
  prep(pre_W, pre_Wt, 1024, 1027, 1056, 1);
  prep(W_ih,  W_ihb,  6144, 1024, 1024, 0);   // used as x @ W.T -> already N x K
  prep(W_hh,  W_hhb,  6144, 2048, 2048, 0);
  prep(tW1,   tW1t,   1024, 2048, 2048, 1);
  prep(tW2,   tW2t,   1024, 1024, 1024, 1);
  prep(rW1,   rW1t,   1024, 3072, 3072, 1);
  prep(rW2,   rW2t,   1024, 1024, 1024, 1);
  prep(dW1,   dW1t,   1024, 3072, 3072, 1);
  prep(dW2,   dW2t,   2048, 1024, 1024, 1);

  // ---- init state + accumulators ----
  zero_kernel<<<dim3((B_ * RECS + 255) / 256), dim3(256), 0, stream>>>(hping, (size_t)B_ * RECS);
  zero_kernel<<<dim3((B_ * 1024 + 255) / 256), dim3(256), 0, stream>>>(lping, (size_t)B_ * 1024);
  zero_kernel<<<dim3(1), dim3(256), 0, stream>>>(accs, 64);

  // ---- encoder: elu(norm(zs) @ enc_W + enc_b), fused normalization ----
  gemm(zs, 2048, enc_Wt, enc_b, encoded, B_ * T_, 1024, 2048, 1, z_mean, z_std);

  // ---- sequential scan, decoder + losses fused per step ----
  float* h_cur = hping; float* h_nxt = hpong;
  float* l_cur = lping; float* l_nxt = lpong;
  for (int t = 0; t < T_ - 1; ++t) {
    concat2(pre_in, 1056, l_cur, 1024, 1024, actions + (size_t)t * 3, 3, T_ * 3);
    gemm(pre_in, 1056, pre_Wt, pre_b, xbuf, B_, 1024, 1056, 1, nullptr, nullptr);
    gemm(xbuf, 1024, W_ihb, b_ih, gxbuf, B_, 6144, 1024, 0, nullptr, nullptr);
    gemm(h_cur, 2048, W_hhb, b_hh, ghbuf, B_, 6144, 2048, 0, nullptr, nullptr);
    gru_kernel<<<dim3(B_ * RECS / 256), dim3(256), 0, stream>>>(gxbuf, ghbuf, h_cur, h_nxt, B_);
    gemm(h_nxt, 2048, tW1t, tb1, t1buf, B_, 1024, 2048, 1, nullptr, nullptr);
    gemm(t1buf, 1024, tW2t, tb2, priorb, B_, 1024, 1024, 0, nullptr, nullptr);
    concat2(post_in, 3072, h_nxt, 2048, 2048, encoded + (size_t)(t + 1) * 1024, 1024, T_ * 1024);
    gemm(post_in, 3072, rW1t, rb1, r1buf, B_, 1024, 3072, 1, nullptr, nullptr);
    gemm(r1buf, 1024, rW2t, rb2, postb, B_, 1024, 1024, 0, nullptr, nullptr);
    latent_kernel<<<dim3(B_ * 32 / 8), dim3(256), 0, stream>>>(postb, l_nxt, B_);
    kl_kernel<<<dim3(B_ / 8), dim3(256), 0, stream>>>(postb, priorb, accs + 1, B_);
    concat2(dec_in, 3072, h_nxt, 2048, 2048, l_nxt, 1024, 1024);
    gemm(dec_in, 3072, dW1t, db1, d1buf, B_, 1024, 3072, 1, nullptr, nullptr);
    gemm(d1buf, 1024, dW2t, db2, predb, B_, 2048, 1024, 0, nullptr, nullptr);
    recon_kernel<<<dim3(B_ * 2048 / 256), dim3(256), 0, stream>>>(predb, zs, z_mean, z_std, accs, t);
    float* tmp;
    tmp = h_cur; h_cur = h_nxt; h_nxt = tmp;
    tmp = l_cur; l_cur = l_nxt; l_nxt = tmp;
  }

  finalize_kernel<<<dim3(1), dim3(1), 0, stream>>>(
      accs, accs + 1, (float*)d_out,
      (float)B_ * (T_ - 1) * 2048.0f, (float)B_ * (T_ - 1));
}
